// MTCNN_19267223289936
// MI455X (gfx1250) — matrix operand order repaired and checked by reading, not compiled, not run on hardware
//
#include <hip/hip_runtime.h>
#include <stdint.h>

typedef __attribute__((ext_vector_type(16))) _Float16 v16h;
typedef __attribute__((ext_vector_type(8)))  float    v8f;

union ABfrag { uint32_t u[8]; v16h h; };

namespace {
constexpr int BATCH = 4;
constexpr int H0v = 1080, W0v = 1920;
constexpr int H1v = 1078, W1v = 1918;
constexpr int HPv = 539,  WPv = 959;
constexpr int H2v = 537,  W2v = 957;
constexpr int H3v = 535,  W3v = 955;
constexpr long long S1 = (long long)H1v * W1v;
constexpr long long SP = (long long)HPv * WPv;
constexpr long long S2 = (long long)H2v * W2v;
constexpr long long S3 = (long long)H3v * W3v;
constexpr long long BN3 = (long long)BATCH * S3;
constexpr int TOPK = 1024;
constexpr int CAP  = 4096;   // candidate buffer for top-k selection
constexpr int NBIN = 16384;  // score histogram bins

constexpr size_t AL(size_t x) { return (x + 255) & ~(size_t)255; }
constexpr size_t OFF_PACK = 0;                                   // 15 frags * 32 lanes * 8 dw
constexpr size_t OFF_HIST = AL(OFF_PACK + (size_t)15*32*8*4);
constexpr size_t OFF_CNT  = AL(OFF_HIST + (size_t)NBIN*4);
constexpr size_t OFF_CS   = AL(OFF_CNT  + 64);
constexpr size_t OFF_CI   = AL(OFF_CS   + (size_t)CAP*4);
constexpr size_t OFF_TB   = AL(OFF_CI   + (size_t)CAP*4);
constexpr size_t OFF_TP   = AL(OFF_TB   + (size_t)TOPK*16);
constexpr size_t OFF_TS   = AL(OFF_TP   + (size_t)TOPK*16);
constexpr size_t OFF_TIMG = AL(OFF_TS   + (size_t)TOPK*4);
constexpr size_t OFF_TVAL = AL(OFF_TIMG + (size_t)TOPK*4);
constexpr size_t OFF_KEEP = AL(OFF_TVAL + (size_t)TOPK*4);
constexpr size_t OFF_H1   = AL(OFF_KEEP + (size_t)TOPK*4);
constexpr size_t OFF_H1P  = AL(OFF_H1  + (size_t)BATCH*10*S1*2);
constexpr size_t OFF_H2   = AL(OFF_H1P + (size_t)BATCH*10*SP*2);
constexpr size_t OFF_H3   = AL(OFF_H2  + (size_t)BATCH*16*S2*2);
constexpr size_t OFF_SC   = AL(OFF_H3  + (size_t)BATCH*32*S3*2);
constexpr size_t OFF_PR   = AL(OFF_SC  + (size_t)BN3*4);
} // namespace

__device__ __forceinline__ uint32_t pack2h(float lo, float hi) {
  _Float16 l = (_Float16)lo, h = (_Float16)hi;
  uint16_t lb = __builtin_bit_cast(uint16_t, l);
  uint16_t hb = __builtin_bit_cast(uint16_t, h);
  return (uint32_t)lb | ((uint32_t)hb << 16);
}

// ---------------------------------------------------------------------------
// Weight packing into 16-bit WMMA A fragments, contiguous k order (same as B):
//   lane L: M = L&15; VGPR j carries K = 2*j + 16*(L>>4) (+half)
// Frags: [0]=conv1, [1..3]=conv2 kt, [4..13]=conv3 (mt*5+kt), [14]=heads.
// K-order is ci*KSZ*KSZ + kh*KSZ + kw == flat OIHW index, so A[m][k]=w[m*K+k].
// ---------------------------------------------------------------------------
__global__ void pack_weights_kernel(const float* __restrict__ w1, const float* __restrict__ w2,
                                    const float* __restrict__ w3, const float* __restrict__ w41,
                                    const float* __restrict__ w42, uint32_t* __restrict__ pack) {
  int tid = blockIdx.x * blockDim.x + threadIdx.x;
  if (tid >= 15 * 32 * 8) return;
  int slot = tid & 7;
  int lane = (tid >> 3) & 31;
  int frag = tid >> 8;
  int hi = lane >> 4, m16 = lane & 15;
  int kbase = slot * 2 + hi * 16;
  int layer, mt, kt;
  if (frag == 0)      { layer = 0; mt = 0; kt = 0; }
  else if (frag < 4)  { layer = 1; mt = 0; kt = frag - 1; }
  else if (frag < 14) { layer = 2; mt = (frag - 4) / 5; kt = (frag - 4) % 5; }
  else                { layer = 3; mt = 0; kt = 0; }
  float v[2] = {0.f, 0.f};
#pragma unroll
  for (int half = 0; half < 2; ++half) {
    int k = kt * 32 + kbase + half;
    int m = mt * 16 + m16;
    float val = 0.f;
    if (layer == 0)      { if (m < 10 && k < 27)  val = w1[m * 27 + k]; }
    else if (layer == 1) { if (m < 16 && k < 90)  val = w2[m * 90 + k]; }
    else if (layer == 2) { if (m < 32 && k < 144) val = w3[m * 144 + k]; }
    else { if (k < 32) { if (m < 2) val = w41[m * 32 + k]; else if (m < 6) val = w42[(m - 2) * 32 + k]; } }
    v[half] = val;
  }
  pack[(size_t)frag * 256 + lane * 8 + slot] = pack2h(v[0], v[1]);
}

// ---------------------------------------------------------------------------
// Implicit-GEMM conv + bias + PReLU with v_wmma_f32_16x16x32_f16.
// One wave = NT tiles of 16 output positions, MT tiles of 16 output channels.
// A fragment loaded once per (kt,mt) and reused across all NT column tiles.
// B fragment (contiguous k order, same as A): lane (hi,n): VGPR j holds K=16*hi+2j(+1).
// ---------------------------------------------------------------------------
template <typename TIN, int CIN, int COUT, int KSZ, int MT, int KT, int NT, bool DO_PRELU>
__global__ __launch_bounds__(256)
void conv_wmma_kernel(const TIN* __restrict__ in, const uint32_t* __restrict__ pack, int fragBase,
                      const float* __restrict__ bias, const float* __restrict__ alpha,
                      _Float16* __restrict__ out, int Bn, int Ho, int Wo, int Hi, int Wi) {
  constexpr int K = CIN * KSZ * KSZ;
  const int lane = threadIdx.x & 31;
  const long long wave = ((long long)blockIdx.x * blockDim.x + threadIdx.x) >> 5;
  const int hi = lane >> 4, nl = lane & 15;
  const long long S = (long long)Ho * Wo;
  const long long Ntot = S * Bn;

  long long pos[NT];
  bool nvalid[NT];
  long long rem[NT];
  int bidx[NT], oy[NT], ox[NT];
  const TIN* inb[NT];
#pragma unroll
  for (int t = 0; t < NT; ++t) {
    long long p = wave * (16 * NT) + t * 16 + nl;
    nvalid[t] = p < Ntot;
    if (!nvalid[t]) p = Ntot - 1;  // clamp; no divergence across WMMA (EXEC all-ones)
    pos[t] = p;
    int b = (int)(p / S);
    long long r = p - (long long)b * S;
    bidx[t] = b;
    rem[t] = r;
    oy[t] = (int)(r / Wo);
    ox[t] = (int)(r - (long long)oy[t] * Wo);
    inb[t] = in + (long long)b * CIN * Hi * Wi;
  }

  v8f zero = {0.f, 0.f, 0.f, 0.f, 0.f, 0.f, 0.f, 0.f};
  v8f acc[MT][NT];
#pragma unroll
  for (int mt = 0; mt < MT; ++mt)
#pragma unroll
    for (int t = 0; t < NT; ++t) acc[mt][t] = zero;

  for (int kt = 0; kt < KT; ++kt) {
    ABfrag bf[NT];
#pragma unroll
    for (int t = 0; t < NT; ++t) {
#pragma unroll
      for (int j = 0; j < 8; ++j) {
        float v0 = 0.f, v1 = 0.f;
        int kb = kt * 32 + hi * 16 + 2 * j;
#pragma unroll
        for (int half = 0; half < 2; ++half) {
          int k = kb + half;
          float v = 0.f;
          if (k < K) {
            int ci, kh, kw;
            if (KSZ == 1) { ci = k; kh = 0; kw = 0; }
            else { ci = k / (KSZ * KSZ); int r = k - ci * (KSZ * KSZ); kh = r / KSZ; kw = r - kh * KSZ; }
            v = (float)inb[t][((long long)ci * Hi + (oy[t] + kh)) * Wi + (ox[t] + kw)];
          }
          if (half) v1 = v; else v0 = v;
        }
        bf[t].u[j] = pack2h(v0, v1);
      }
    }
#pragma unroll
    for (int mt = 0; mt < MT; ++mt) {
      ABfrag af;
      const uint4* ap = (const uint4*)(pack + (((size_t)(fragBase + mt * KT + kt) * 32 + lane) * 8));
      uint4 q0 = ap[0], q1 = ap[1];
      af.u[0] = q0.x; af.u[1] = q0.y; af.u[2] = q0.z; af.u[3] = q0.w;
      af.u[4] = q1.x; af.u[5] = q1.y; af.u[6] = q1.z; af.u[7] = q1.w;
#pragma unroll
      for (int t = 0; t < NT; ++t) {
        acc[mt][t] = __builtin_amdgcn_wmma_f32_16x16x32_f16(false, af.h, false, bf[t].h, (short)0,
                                                            acc[mt][t], false, false);
      }
    }
  }
  // D layout: lane (hi,n): VGPR r holds M = r + 8*hi, N = n.
#pragma unroll
  for (int t = 0; t < NT; ++t) {
    if (!nvalid[t]) continue;
#pragma unroll
    for (int mt = 0; mt < MT; ++mt) {
#pragma unroll
      for (int r = 0; r < 8; ++r) {
        int m = mt * 16 + hi * 8 + r;
        if (m < COUT) {
          float v = acc[mt][t][r] + bias[m];
          if (DO_PRELU) v = (v >= 0.f) ? v : alpha[m] * v;
          out[((long long)bidx[t] * COUT + m) * S + rem[t]] = (_Float16)v;
        }
      }
    }
  }
}

// 2x2/2 max-pool over f16 planes (H1,W1 are even -> exact).
__global__ void maxpool_kernel(const _Float16* __restrict__ in, _Float16* __restrict__ out,
                               int Hi, int Wi, int Ho, int Wo, long long total) {
  long long i = (long long)blockIdx.x * blockDim.x + threadIdx.x;
  if (i >= total) return;
  long long so = (long long)Ho * Wo;
  long long plane = i / so;
  long long rem = i - plane * so;
  int ho = (int)(rem / Wo), wo = (int)(rem - (long long)ho * Wo);
  const _Float16* p = in + plane * (long long)Hi * Wi;
  int y = 2 * ho, x = 2 * wo;
  float v = (float)p[(long long)y * Wi + x];
  if (x + 1 < Wi) v = fmaxf(v, (float)p[(long long)y * Wi + x + 1]);
  if (y + 1 < Hi) {
    v = fmaxf(v, (float)p[(long long)(y + 1) * Wi + x]);
    if (x + 1 < Wi) v = fmaxf(v, (float)p[(long long)(y + 1) * Wi + x + 1]);
  }
  out[i] = (_Float16)v;
}

// Fused 1x1 heads: one WMMA (rows 0-1 logits -> softmax prob, rows 2-5 bbox pred).
__global__ __launch_bounds__(256)
void head_kernel(const _Float16* __restrict__ h3, const uint32_t* __restrict__ pack, int fragBase,
                 const float* __restrict__ b41, const float* __restrict__ b42,
                 float* __restrict__ scores, float* __restrict__ pred, int Bn, int Ho, int Wo) {
  const int lane = threadIdx.x & 31;
  const long long wave = ((long long)blockIdx.x * blockDim.x + threadIdx.x) >> 5;
  const int hi = lane >> 4, nl = lane & 15;
  const long long S = (long long)Ho * Wo;
  const long long Ntot = S * Bn;
  long long pos = wave * 16 + nl;
  const bool nvalid = pos < Ntot;
  if (!nvalid) pos = Ntot - 1;
  const int b = (int)(pos / S);
  const long long rem = pos - (long long)b * S;
  const _Float16* inb = h3 + (long long)b * 32 * S;
  ABfrag bf;
#pragma unroll
  for (int j = 0; j < 8; ++j) {
    int k0 = hi * 16 + 2 * j;
    float v0 = (float)inb[(long long)k0 * S + rem];
    float v1 = (float)inb[(long long)(k0 + 1) * S + rem];
    bf.u[j] = pack2h(v0, v1);
  }
  ABfrag af;
  const uint4* ap = (const uint4*)(pack + (((size_t)fragBase * 32 + lane) * 8));
  uint4 q0 = ap[0], q1 = ap[1];
  af.u[0] = q0.x; af.u[1] = q0.y; af.u[2] = q0.z; af.u[3] = q0.w;
  af.u[4] = q1.x; af.u[5] = q1.y; af.u[6] = q1.z; af.u[7] = q1.w;
  v8f acc = {0.f, 0.f, 0.f, 0.f, 0.f, 0.f, 0.f, 0.f};
  acc = __builtin_amdgcn_wmma_f32_16x16x32_f16(false, af.h, false, bf.h, (short)0, acc, false, false);
  if (nvalid && hi == 0) {  // rows 0..7 live in lanes 0..15
    float l0 = acc[0] + b41[0];
    float l1 = acc[1] + b41[1];
    float mx = fmaxf(l0, l1);
    float e0 = __expf(l0 - mx), e1 = __expf(l1 - mx);
    scores[pos] = e1 / (e0 + e1);
#pragma unroll
    for (int c = 0; c < 4; ++c) pred[pos * 4 + c] = acc[2 + c] + b42[c];
  }
}

// ------------------------- top-k via histogram select -----------------------
__global__ void zero_hist_kernel(unsigned int* __restrict__ hist) {
  int i = blockIdx.x * blockDim.x + threadIdx.x;
  if (i < NBIN) hist[i] = 0u;
}

__global__ void hist_kernel(const float* __restrict__ scores, unsigned int* __restrict__ hist, long long n) {
  long long i = (long long)blockIdx.x * blockDim.x + threadIdx.x;
  if (i >= n) return;
  float s = scores[i];
  if (s >= 0.6f) {
    int b = (int)(s * 16384.f);
    if (b > NBIN - 1) b = NBIN - 1;
    if (b < 0) b = 0;
    atomicAdd(&hist[b], 1u);
  }
}

__global__ void cutoff_kernel(const unsigned int* __restrict__ hist, unsigned int* __restrict__ cnt) {
  if (threadIdx.x == 0 && blockIdx.x == 0) {
    long long s = 0;
    int c = 0;
    for (int b = NBIN - 1; b >= 0; --b) {
      s += hist[b];
      if (s >= (long long)TOPK) { c = b; break; }
    }
    cnt[1] = (unsigned)c;
    cnt[0] = 0u;  // reset compaction counter
  }
}

__global__ void topinit_kernel(float* __restrict__ tb, float* __restrict__ tp, float* __restrict__ ts,
                               int* __restrict__ timg, int* __restrict__ tval) {
  int t = threadIdx.x;
  ts[t] = -1e30f; timg[t] = 0; tval[t] = 0;
#pragma unroll
  for (int c = 0; c < 4; ++c) { tb[t * 4 + c] = 0.f; tp[t * 4 + c] = 0.f; }
}

__global__ void compact_kernel(const float* __restrict__ scores, unsigned int* __restrict__ cnt,
                               float* __restrict__ cs, unsigned int* __restrict__ ci, long long n) {
  long long i = (long long)blockIdx.x * blockDim.x + threadIdx.x;
  if (i >= n) return;
  float s = scores[i];
  if (s >= 0.6f) {
    int b = (int)(s * 16384.f);
    if (b > NBIN - 1) b = NBIN - 1;
    if (b < 0) b = 0;
    if (b >= (int)cnt[1]) {
      unsigned slot = atomicAdd(&cnt[0], 1u);
      if (slot < (unsigned)CAP) { cs[slot] = s; ci[slot] = (unsigned)i; }
    }
  }
}

// Exact rank selection among <=CAP candidates: (score desc, idx asc) == lax.top_k order.
__global__ __launch_bounds__(1024)
void topsel_kernel(const float* __restrict__ cs, const unsigned int* __restrict__ ci,
                   const unsigned int* __restrict__ cnt, const float* __restrict__ pred,
                   float* __restrict__ tb, float* __restrict__ tp, float* __restrict__ ts,
                   int* __restrict__ timg, int* __restrict__ tval) {
  __shared__ float ss[CAP];
  __shared__ unsigned int si[CAP];
  int n = (int)cnt[0];
  if (n > CAP) n = CAP;
  for (int t = threadIdx.x; t < CAP; t += 1024) {
    ss[t] = (t < n) ? cs[t] : -1e30f;
    si[t] = (t < n) ? ci[t] : 0xFFFFFFFFu;
  }
  __syncthreads();
  for (int t = threadIdx.x; t < CAP; t += 1024) {
    if (t >= n) continue;
    float s = ss[t];
    unsigned id = si[t];
    int rank = 0;
    for (int j = 0; j < CAP; ++j) {
      float sj = ss[j];
      unsigned ij = si[j];
      rank += ((sj > s) || (sj == s && ij < id)) ? 1 : 0;
    }
    if (rank < TOPK) {
      long long gid = (long long)id;
      int img = (int)(gid / S3);
      long long p = gid - (long long)img * S3;
      int oy = (int)(p / W3v), ox = (int)(p - (long long)oy * W3v);
      tb[rank * 4 + 0] = floorf(2.f * ox + 1.f);
      tb[rank * 4 + 1] = floorf(2.f * oy + 1.f);
      tb[rank * 4 + 2] = floorf(2.f * ox + 12.f);
      tb[rank * 4 + 3] = floorf(2.f * oy + 12.f);
#pragma unroll
      for (int c = 0; c < 4; ++c) tp[rank * 4 + c] = pred[gid * 4 + c];
      ts[rank] = s; timg[rank] = img; tval[rank] = 1;
    }
  }
}

// Greedy NMS, one 1024-thread block (sequential over boxes, parallel over IoU).
__global__ __launch_bounds__(1024)
void nms_kernel(const float* __restrict__ tb, const int* __restrict__ timg,
                const int* __restrict__ tval, int* __restrict__ keep) {
  __shared__ float bx1[TOPK], by1[TOPK], bx2[TOPK], by2[TOPK], ar[TOPK];
  __shared__ int kp[TOPK];
  int t = threadIdx.x;
  float off = (float)timg[t] * 100000.f;
  float x1 = tb[t * 4 + 0] + off, y1 = tb[t * 4 + 1] + off;
  float x2 = tb[t * 4 + 2] + off, y2 = tb[t * 4 + 3] + off;
  bx1[t] = x1; by1[t] = y1; bx2[t] = x2; by2[t] = y2;
  ar[t] = (x2 - x1) * (y2 - y1);
  kp[t] = tval[t];
  __syncthreads();
  for (int i = 0; i < TOPK; ++i) {
    if (kp[i]) {  // uniform branch across block
      if (t > i) {
        float xx1 = fmaxf(bx1[i], x1), yy1 = fmaxf(by1[i], y1);
        float xx2 = fminf(bx2[i], x2), yy2 = fminf(by2[i], y2);
        float inter = fmaxf(xx2 - xx1, 0.f) * fmaxf(yy2 - yy1, 0.f);
        float iou = inter / (ar[i] + ar[t] - inter);
        if (iou > 0.5f) kp[t] = 0;
      }
    }
    __syncthreads();
  }
  keep[t] = kp[t];
}

__global__ void finalize_kernel(const float* __restrict__ tb, const float* __restrict__ tp,
                                const float* __restrict__ ts, const int* __restrict__ keep,
                                float* __restrict__ out) {
  int t = blockIdx.x * blockDim.x + threadIdx.x;
  if (t >= TOPK) return;
  float x1 = tb[t * 4 + 0], y1 = tb[t * 4 + 1], x2 = tb[t * 4 + 2], y2 = tb[t * 4 + 3];
  float w = x2 - x1, h = y2 - y1;
  float rx1 = x1 + tp[t * 4 + 0] * w, ry1 = y1 + tp[t * 4 + 1] * h;
  float rx2 = x2 + tp[t * 4 + 2] * w, ry2 = y2 + tp[t * 4 + 3] * h;
  float rw = rx2 - rx1, rh = ry2 - ry1, l = fmaxf(rw, rh);
  float sx1 = rx1 + 0.5f * rw - 0.5f * l, sy1 = ry1 + 0.5f * rh - 0.5f * l;
  bool k = keep[t] != 0;
  out[t * 5 + 0] = k ? sx1 : 0.f;
  out[t * 5 + 1] = k ? sy1 : 0.f;
  out[t * 5 + 2] = k ? (sx1 + l) : 0.f;
  out[t * 5 + 3] = k ? (sy1 + l) : 0.f;
  out[t * 5 + 4] = k ? ts[t] : 0.f;
}

extern "C" void kernel_launch(void* const* d_in, const int* in_sizes, int n_in,
                              void* d_out, int out_size, void* d_ws, size_t ws_size,
                              hipStream_t stream) {
  (void)in_sizes; (void)n_in; (void)out_size; (void)ws_size;
  const float* x   = (const float*)d_in[0];
  const float* w1  = (const float*)d_in[1];
  const float* b1  = (const float*)d_in[2];
  const float* a1  = (const float*)d_in[3];
  const float* w2  = (const float*)d_in[4];
  const float* b2  = (const float*)d_in[5];
  const float* a2  = (const float*)d_in[6];
  const float* w3  = (const float*)d_in[7];
  const float* b3  = (const float*)d_in[8];
  const float* a3  = (const float*)d_in[9];
  const float* w41 = (const float*)d_in[10];
  const float* b41 = (const float*)d_in[11];
  const float* w42 = (const float*)d_in[12];
  const float* b42 = (const float*)d_in[13];
  // d_in[14] is k (==1024, compile-time TOPK here).

  char* ws = (char*)d_ws;
  uint32_t*     pack   = (uint32_t*)(ws + OFF_PACK);
  unsigned int* hist   = (unsigned int*)(ws + OFF_HIST);
  unsigned int* cnt    = (unsigned int*)(ws + OFF_CNT);
  float*        cs     = (float*)(ws + OFF_CS);
  unsigned int* ci     = (unsigned int*)(ws + OFF_CI);
  float*        tb     = (float*)(ws + OFF_TB);
  float*        tp     = (float*)(ws + OFF_TP);
  float*        tsv    = (float*)(ws + OFF_TS);
  int*          timg   = (int*)(ws + OFF_TIMG);
  int*          tval   = (int*)(ws + OFF_TVAL);
  int*          keep   = (int*)(ws + OFF_KEEP);
  _Float16*     h1     = (_Float16*)(ws + OFF_H1);
  _Float16*     h1p    = (_Float16*)(ws + OFF_H1P);
  _Float16*     h2     = (_Float16*)(ws + OFF_H2);
  _Float16*     h3     = (_Float16*)(ws + OFF_H3);
  float*        scores = (float*)(ws + OFF_SC);
  float*        pred   = (float*)(ws + OFF_PR);

  // NT=2 column tiles per wave -> 32 output positions per wave.
  auto nblk2 = [](long long ntot) {
    long long waves = (ntot + 31) / 32;
    long long thr = waves * 32;
    return (unsigned)((thr + 255) / 256);
  };
  auto nblk1 = [](long long ntot) {
    long long waves = (ntot + 15) / 16;
    long long thr = waves * 32;
    return (unsigned)((thr + 255) / 256);
  };

  pack_weights_kernel<<<15, 256, 0, stream>>>(w1, w2, w3, w41, w42, pack);

  conv_wmma_kernel<float, 3, 10, 3, 1, 1, 2, true>
      <<<nblk2(BATCH * S1), 256, 0, stream>>>(x, pack, 0, b1, a1, h1, BATCH, H1v, W1v, H0v, W0v);

  long long poolN = (long long)BATCH * 10 * SP;
  maxpool_kernel<<<(unsigned)((poolN + 255) / 256), 256, 0, stream>>>(h1, h1p, H1v, W1v, HPv, WPv, poolN);

  conv_wmma_kernel<_Float16, 10, 16, 3, 1, 3, 2, true>
      <<<nblk2(BATCH * S2), 256, 0, stream>>>(h1p, pack, 1, b2, a2, h2, BATCH, H2v, W2v, HPv, WPv);

  conv_wmma_kernel<_Float16, 16, 32, 3, 2, 5, 2, true>
      <<<nblk2(BATCH * S3), 256, 0, stream>>>(h2, pack, 4, b3, a3, h3, BATCH, H3v, W3v, H2v, W2v);

  head_kernel<<<nblk1(BN3), 256, 0, stream>>>(h3, pack, 14, b41, b42, scores, pred, BATCH, H3v, W3v);

  zero_hist_kernel<<<NBIN / 256, 256, 0, stream>>>(hist);
  hist_kernel<<<(unsigned)((BN3 + 255) / 256), 256, 0, stream>>>(scores, hist, BN3);
  cutoff_kernel<<<1, 32, 0, stream>>>(hist, cnt);
  topinit_kernel<<<1, TOPK, 0, stream>>>(tb, tp, tsv, timg, tval);
  compact_kernel<<<(unsigned)((BN3 + 255) / 256), 256, 0, stream>>>(scores, cnt, cs, ci, BN3);
  topsel_kernel<<<1, 1024, 0, stream>>>(cs, ci, cnt, pred, tb, tp, tsv, timg, tval);
  nms_kernel<<<1, TOPK, 0, stream>>>(tb, timg, tval, keep);
  finalize_kernel<<<(TOPK + 255) / 256, 256, 0, stream>>>(tb, tp, tsv, keep, (float*)d_out);
}
